// Atten_Head_78099685311047
// MI455X (gfx1250) — matrix operand order repaired and checked by reading, compiled, not run on hardware
//
#include <hip/hip_runtime.h>
#include <hip/hip_bf16.h>

// ---------------------------------------------------------------------------
// Types for CDNA5 WMMA (wave32): v_wmma_f32_16x16x32_bf16
// A: 16x32 bf16 (8 VGPRs), B: 32x16 bf16 (8 VGPRs), C/D: 16x16 f32 (8 VGPRs)
// ---------------------------------------------------------------------------
typedef __bf16 bf16_t;
typedef __attribute__((ext_vector_type(8)))  bf16_t v8bf;
typedef __attribute__((ext_vector_type(16))) bf16_t v16bf;
typedef __attribute__((ext_vector_type(8)))  float  v8f;
typedef __attribute__((ext_vector_type(4)))  float  v4f;
typedef int v4i_gcc __attribute__((vector_size(16)));

#define BATCH 8
#define SEQ   2048
#define DIM   1024
#define KROW  264            // padded LDS row (32 keys x 256 dims + 8 pad), 528B = 33*16B

__device__ __forceinline__ v8f vzero8() {
  v8f r;
#pragma unroll
  for (int i = 0; i < 8; ++i) r[i] = 0.0f;
  return r;
}

// Two 16-byte bf16 chunks -> one 16-element WMMA operand register set.
// A operand: ld_pair(p, p+16)  (runs of 8, second run 16 elements later)
// B operand: ld_pair(p, p+16)  (same two runs of 8 as A)
__device__ __forceinline__ v16bf ld_pair(const bf16_t* p0, const bf16_t* p1) {
  v8bf a = *(const v8bf*)p0;
  v8bf b = *(const v8bf*)p1;
  v16bf r;
#pragma unroll
  for (int i = 0; i < 8; ++i) { r[i] = a[i]; r[i + 8] = b[i]; }
  return r;
}

// A operand straight from fp32 memory (two runs of 8 floats), cvt to bf16.
__device__ __forceinline__ v16bf ld_a_f32(const float* p0, const float* p1) {
  v4f a0 = *(const v4f*)p0, a1 = *(const v4f*)(p0 + 4);
  v4f b0 = *(const v4f*)p1, b1 = *(const v4f*)(p1 + 4);
  v16bf r;
#pragma unroll
  for (int i = 0; i < 4; ++i) {
    r[i]      = (bf16_t)a0[i];
    r[i + 4]  = (bf16_t)a1[i];
    r[i + 8]  = (bf16_t)b0[i];
    r[i + 12] = (bf16_t)b1[i];
  }
  return r;
}

__device__ __forceinline__ v8f wmma_bf16(v16bf a, v16bf b, v8f c) {
  // (neg_a, A, neg_b, B, c_mod, C, reuse_a, reuse_b)
  return __builtin_amdgcn_wmma_f32_16x16x32_bf16(false, a, false, b,
                                                 (short)0, c, false, false);
}

__device__ __forceinline__ float rmax16(float v) {
#pragma unroll
  for (int o = 1; o < 16; o <<= 1) v = fmaxf(v, __shfl_xor(v, o, 32));
  return v;
}
__device__ __forceinline__ float rsum16(float v) {
#pragma unroll
  for (int o = 1; o < 16; o <<= 1) v += __shfl_xor(v, o, 32);
  return v;
}

// ---------------------------------------------------------------------------
// Async global->LDS copy (ASYNCcnt path) + wait.
// Builtin first param is a generic v4i* (per hipcc diagnostic); LDS param is
// cast to AS(3) v4i* (exact if AS3-typed, implicitly widened if generic).
// ---------------------------------------------------------------------------
__device__ __forceinline__ void async_cp_b128(const bf16_t* g, bf16_t* l) {
#if __has_builtin(__builtin_amdgcn_global_load_async_to_lds_b128)
  __builtin_amdgcn_global_load_async_to_lds_b128(
      (v4i_gcc*)g,
      (__attribute__((address_space(3))) v4i_gcc*)(v4i_gcc*)l,
      0, 0);
#else
  asm volatile("global_load_async_to_lds_b128 %0, %1, off"
               :: "v"((unsigned)(unsigned long long)l),
                  "v"(g)
               : "memory");
#endif
}

// Wait until ASYNCcnt <= n, plus a compiler memory barrier so LDS reads of
// staged data are not hoisted above the wait.
#if __has_builtin(__builtin_amdgcn_s_wait_asynccnt)
#define WAIT_ASYNCCNT(n)                         \
  do {                                           \
    __builtin_amdgcn_s_wait_asynccnt(n);         \
    asm volatile("" ::: "memory");               \
  } while (0)
#else
#define WAIT_ASYNCCNT(n) asm volatile("s_wait_asynccnt %0" :: "n"(n) : "memory")
#endif

// ---------------------------------------------------------------------------
// Kernel 1: fp32 weights -> bf16 transposed weights WT[n][k] = W[k][n]
// ---------------------------------------------------------------------------
__global__ void cvt_weights(const float* __restrict__ Kw,
                            const float* __restrict__ Vw,
                            const float* __restrict__ Qw,
                            bf16_t* __restrict__ WTq,
                            bf16_t* __restrict__ WTk,
                            bf16_t* __restrict__ WTv) {
  int t = blockIdx.x * 256 + threadIdx.x;      // 0 .. 1024*1024-1
  int z = blockIdx.y;                          // 0:Q 1:K 2:V
  int o = t >> 10;                             // output (n) index
  int i = t & 1023;                            // input (k) index -> coalesced writes
  const float* W  = (z == 0) ? Qw  : (z == 1) ? Kw  : Vw;
  bf16_t*      WT = (z == 0) ? WTq : (z == 1) ? WTk : WTv;
  WT[(size_t)o * DIM + i] = (bf16_t)W[(size_t)i * DIM + o];
}

// ---------------------------------------------------------------------------
// Kernel 2: projections. X[B*S,1024](f32) x WT(bf16) -> Qb,Kb row-major bf16,
// V stored transposed: VT[b][d][s] bf16.
// Block = 4 waves; wave = 16(M) x 64(N) tile; block = 64x64.
// ---------------------------------------------------------------------------
__global__ __launch_bounds__(128, 1)
void proj_kernel(const float* __restrict__ X,
                 const bf16_t* __restrict__ WTq,
                 const bf16_t* __restrict__ WTk,
                 const bf16_t* __restrict__ WTv,
                 bf16_t* __restrict__ Qb,
                 bf16_t* __restrict__ Kb,
                 bf16_t* __restrict__ VTb) {
  const int lane  = threadIdx.x & 31;
  const int wv    = threadIdx.x >> 5;
  const int col16 = lane & 15;
  const int half  = lane >> 4;
  const int n0 = blockIdx.x * 64;
  const int m0 = blockIdx.y * 64 + wv * 16;
  const int z  = blockIdx.z;
  const bf16_t* WT = (z == 0) ? WTq : (z == 1) ? WTk : WTv;

  v8f acc[4];
#pragma unroll
  for (int nt = 0; nt < 4; ++nt) acc[nt] = vzero8();

  const float* xrow_base = X + (size_t)(m0 + col16) * DIM;
  for (int k = 0; k < DIM; k += 32) {
    const float* xp = xrow_base + k + half * 8;
    v16bf a = ld_a_f32(xp, xp + 16);
#pragma unroll
    for (int nt = 0; nt < 4; ++nt) {
      const bf16_t* bp = WT + (size_t)(n0 + nt * 16 + col16) * DIM + k + half * 8;
      acc[nt] = wmma_bf16(a, ld_pair(bp, bp + 16), acc[nt]);
    }
  }

#pragma unroll
  for (int nt = 0; nt < 4; ++nt) {
#pragma unroll
    for (int e = 0; e < 8; ++e) {
      int M   = e + half * 8;
      int row = m0 + M;                 // flat (b*S + s)
      int col = n0 + nt * 16 + col16;   // output dim
      bf16_t v = (bf16_t)acc[nt][e];
      if (z == 0)      Qb[(size_t)row * DIM + col] = v;
      else if (z == 1) Kb[(size_t)row * DIM + col] = v;
      else {
        int b = row >> 11, s = row & (SEQ - 1);
        VTb[((size_t)b * DIM + col) * SEQ + s] = v;   // V transposed
      }
    }
  }
}

// ---------------------------------------------------------------------------
// Kernel 3: fused causal flash attention.
// Block = 4 waves, one 16-row query tile per block.
// Score D-reduction split over waves (256 dims each), summed via LDS.
// Each wave owns a 256-wide output slice (16 f32 accumulator tiles).
// K chunks are double-buffered into LDS with GLOBAL_LOAD_ASYNC_TO_LDS_B128,
// synchronized per-wave with s_wait_asynccnt (in-order completion).
// ---------------------------------------------------------------------------
__global__ __launch_bounds__(128, 1)
void flash_kernel(const bf16_t* __restrict__ Qb,
                  const bf16_t* __restrict__ Kb,
                  const bf16_t* __restrict__ VTb,
                  float* __restrict__ Out) {
  // 2 buffers x 4 waves x 32 keys x KROW bf16 = 135168 B
  __shared__ __attribute__((aligned(16))) bf16_t sK[2 * 4 * 32 * KROW];
  __shared__ __attribute__((aligned(16))) float  sPart[4 * 2 * 256];  // 8 KB
  __shared__ __attribute__((aligned(16))) bf16_t sP[16 * 32];         // 1 KB

  const int lane  = threadIdx.x & 31;
  const int wv    = threadIdx.x >> 5;
  const int col16 = lane & 15;
  const int half  = lane >> 4;
  const int b     = blockIdx.y;
  const int q0    = blockIdx.x * 16;
  const int d0    = wv * 256;             // this wave's output / reduction slice

  const bf16_t* Qbase = Qb  + (size_t)b * SEQ * DIM;
  const bf16_t* Kbase = Kb  + (size_t)b * SEQ * DIM;
  const bf16_t* Vt    = VTb + (size_t)b * DIM * SEQ;

  bf16_t* myK0 = sK + (size_t)wv * 32 * KROW;
  bf16_t* myK1 = myK0 + (size_t)4 * 32 * KROW;

  // Preload this wave's Q slice: 8 A-operand chunks of 16x32.
  v16bf qreg[8];
#pragma unroll
  for (int kk = 0; kk < 8; ++kk) {
    const bf16_t* p = Qbase + (size_t)(q0 + col16) * DIM + d0 + kk * 32 + half * 8;
    qreg[kk] = ld_pair(p, p + 16);
  }

  v8f O[16];
#pragma unroll
  for (int dt = 0; dt < 16; ++dt) O[dt] = vzero8();
  float m[8], l[8];
#pragma unroll
  for (int e = 0; e < 8; ++e) { m[e] = -1e30f; l[e] = 0.0f; }

  const int nchunk = (q0 + 16 + 31) >> 5;   // causal: keys 0 .. q0+15

  // Stage chunk 0 of K (this wave's 256-dim slice): 32 rows x 512B.
#pragma unroll
  for (int r = 0; r < 32; ++r)
    async_cp_b128(Kbase + (size_t)r * DIM + d0 + lane * 8,
                  myK0 + r * KROW + lane * 8);

  for (int c = 0; c < nchunk; ++c) {
    const int k0 = c * 32;
    const bool hasnext = (c + 1 < nchunk);
    bf16_t* cur = (c & 1) ? myK1 : myK0;
    bf16_t* nxt = (c & 1) ? myK0 : myK1;

    // Issue async copies for the next chunk, then wait for the current one.
    if (hasnext) {
#pragma unroll
      for (int r = 0; r < 32; ++r)
        async_cp_b128(Kbase + (size_t)(k0 + 32 + r) * DIM + d0 + lane * 8,
                      nxt + r * KROW + lane * 8);
      __builtin_prefetch(Vt + (size_t)(d0 + col16) * SEQ + k0 + 32, 0, 0);
      WAIT_ASYNCCNT(32);   // the 32 copies of chunk c are done (in-order)
    } else {
      WAIT_ASYNCCNT(0);
    }

    // ---- partial scores over this wave's 256-dim slice: two 16x16 tiles ----
    v8f s0 = vzero8(), s1 = vzero8();
#pragma unroll
    for (int kk = 0; kk < 8; ++kk) {
      const bf16_t* kp0 = cur + (col16)      * KROW + kk * 32 + half * 8;
      const bf16_t* kp1 = cur + (16 + col16) * KROW + kk * 32 + half * 8;
      s0 = wmma_bf16(qreg[kk], ld_pair(kp0, kp0 + 16), s0);
      s1 = wmma_bf16(qreg[kk], ld_pair(kp1, kp1 + 16), s1);
    }
#pragma unroll
    for (int e = 0; e < 8; ++e) {
      int M = e + half * 8;
      sPart[wv * 512 +       M * 16 + col16] = s0[e];
      sPart[wv * 512 + 256 + M * 16 + col16] = s1[e];
    }
    __syncthreads();

    // ---- full scores + online softmax (computed redundantly by all waves) --
    float fac[8], p0v[8], p1v[8];
#pragma unroll
    for (int e = 0; e < 8; ++e) {
      int M = e + half * 8;
      float t0 = 0.0f, t1 = 0.0f;
#pragma unroll
      for (int w = 0; w < 4; ++w) {
        t0 += sPart[w * 512 +       M * 16 + col16];
        t1 += sPart[w * 512 + 256 + M * 16 + col16];
      }
      t0 *= 0.03125f;                      // 1/sqrt(1024)
      t1 *= 0.03125f;
      int qq = q0 + M;
      if (k0 + col16 > qq)      t0 = -1e30f;   // causal mask
      if (k0 + 16 + col16 > qq) t1 = -1e30f;
      float mt = fmaxf(rmax16(t0), rmax16(t1));
      float mn = fmaxf(m[e], mt);
      float p0 = __expf(t0 - mn);
      float p1 = __expf(t1 - mn);
      float rs = rsum16(p0) + rsum16(p1);
      float f  = __expf(m[e] - mn);
      l[e] = l[e] * f + rs;
      m[e] = mn;
      fac[e] = f; p0v[e] = p0; p1v[e] = p1;
    }

    // wave 0 deposits P (bf16) row-major for A-layout reload.
    if (wv == 0) {
#pragma unroll
      for (int e = 0; e < 8; ++e) {
        int M = e + half * 8;
        sP[M * 32 +      col16] = (bf16_t)p0v[e];
        sP[M * 32 + 16 + col16] = (bf16_t)p1v[e];
      }
    }
    __syncthreads();

    // ---- O = O*fac + P @ V  on this wave's 256-wide output slice ----------
    const bf16_t* pp = &sP[col16 * 32 + half * 8];
    v16bf aP = ld_pair(pp, pp + 16);

    // Register-rotated pipeline: load V tile dt+1 while WMMA consumes tile dt.
    const bf16_t* vp0 = Vt + (size_t)(d0 + col16) * SEQ + k0 + half * 8;
    v16bf bv = ld_pair(vp0, vp0 + 16);
#pragma unroll
    for (int dt = 0; dt < 16; ++dt) {
      v16bf nbv = bv;
      if (dt < 15) {
        const bf16_t* vpn =
            Vt + (size_t)(d0 + (dt + 1) * 16 + col16) * SEQ + k0 + half * 8;
        nbv = ld_pair(vpn, vpn + 16);
      }
#pragma unroll
      for (int e = 0; e < 8; ++e) O[dt][e] *= fac[e];
      O[dt] = wmma_bf16(aP, bv, O[dt]);
      bv = nbv;
    }
  }

  // ---- epilogue: divide by row sums, store fp32 ----------------------------
  float invl[8];
#pragma unroll
  for (int e = 0; e < 8; ++e) invl[e] = 1.0f / l[e];
  float* ob = Out + (size_t)b * SEQ * DIM;
#pragma unroll
  for (int dt = 0; dt < 16; ++dt) {
#pragma unroll
    for (int e = 0; e < 8; ++e) {
      int M = e + half * 8;
      ob[(size_t)(q0 + M) * DIM + d0 + dt * 16 + col16] = O[dt][e] * invl[e];
    }
  }
}

// ---------------------------------------------------------------------------
extern "C" void kernel_launch(void* const* d_in, const int* in_sizes, int n_in,
                              void* d_out, int out_size, void* d_ws, size_t ws_size,
                              hipStream_t stream) {
  (void)in_sizes; (void)n_in; (void)out_size; (void)ws_size;
  const float* X  = (const float*)d_in[0];   // inputs_for_keys (K,V,Q all from it)
  const float* Kw = (const float*)d_in[3];
  const float* Vw = (const float*)d_in[4];
  const float* Qw = (const float*)d_in[5];
  float* Out = (float*)d_out;

  // Workspace layout (bf16): 3 x WT (2MB) + Qb + Kb + VTb (32MB each) = 102MB
  bf16_t* WTq = (bf16_t*)d_ws;
  bf16_t* WTk = WTq + (size_t)DIM * DIM;
  bf16_t* WTv = WTk + (size_t)DIM * DIM;
  bf16_t* Qb  = WTv + (size_t)DIM * DIM;
  bf16_t* Kb  = Qb  + (size_t)BATCH * SEQ * DIM;
  bf16_t* VTb = Kb  + (size_t)BATCH * SEQ * DIM;

  cvt_weights<<<dim3((DIM * DIM) / 256, 3), 256, 0, stream>>>(Kw, Vw, Qw, WTq, WTk, WTv);
  proj_kernel<<<dim3(DIM / 64, (BATCH * SEQ) / 64, 3), 128, 0, stream>>>(
      X, WTq, WTk, WTv, Qb, Kb, VTb);
  flash_kernel<<<dim3(SEQ / 16, BATCH), 128, 0, stream>>>(Qb, Kb, VTb, Out);
}
